// TextGenerationModel_11708080849327
// MI455X (gfx1250) — matrix operand order repaired and checked by reading, not compiled, not run on hardware
//
#include <hip/hip_runtime.h>
#include <hip/hip_bf16.h>
#include <stdint.h>

#define B_ 256
#define S_ 1024
#define H_ 1024
#define V_ 50257
#define NT_V ((V_ + 15) / 16)   // 3142 column tiles for the projection

typedef __attribute__((ext_vector_type(16))) __bf16 v16bf;
typedef __attribute__((ext_vector_type(8)))  __bf16 v8bf;
typedef __attribute__((ext_vector_type(8)))  float  v8f;

// ---------------- device helpers ----------------

__device__ __forceinline__ unsigned short f2bf(float f) {
  union { float f; unsigned u; } cv; cv.f = f;
  unsigned u = cv.u;
  unsigned r = u + 0x7FFFu + ((u >> 16) & 1u);   // round-to-nearest-even
  return (unsigned short)(r >> 16);
}

__device__ __forceinline__ v16bf cat16(v8bf lo, v8bf hi) {
  v16bf r;
#pragma unroll
  for (int i = 0; i < 8; ++i) { r[i] = lo[i]; r[i + 8] = hi[i]; }
  return r;
}

// A fragment (16x32 bf16, MxK) from row-major bf16 [rows x lda].
// Lanes 0-15 -> M=lane, K = {k0..k0+7, k0+16..k0+23};
// lanes 16-31 -> M=lane-16, K = {k0+8..k0+15, k0+24..k0+31}.
__device__ __forceinline__ v16bf load_a(const __bf16* A, int lda, int mbase,
                                        int k0, int lane) {
  const __bf16* p = A + (size_t)(mbase + (lane & 15)) * lda + k0 + ((lane >> 4) << 3);
  v8bf lo = *(const v8bf*)p;
  v8bf hi = *(const v8bf*)(p + 16);
  return cat16(lo, hi);
}

// B fragment (32x16 bf16, KxN) from pre-packed tiles: each lane reads its own
// 16 contiguous bf16.
__device__ __forceinline__ v16bf load_b(const __bf16* Wp, size_t tile_off, int lane) {
  const __bf16* p = Wp + tile_off + (size_t)lane * 16;
  v8bf lo = *(const v8bf*)p;
  v8bf hi = *(const v8bf*)(p + 8);
  return cat16(lo, hi);
}

__device__ __forceinline__ v8f wmma_bf16(v16bf a, v16bf b, v8f c) {
  return __builtin_amdgcn_wmma_f32_16x16x32_bf16(false, a, false, b, (short)0, c,
                                                 false, false);
}

__device__ __forceinline__ float sigm(float x) { return 1.0f / (1.0f + __expf(-x)); }
__device__ __forceinline__ float tanh_f(float x) { return 2.0f * sigm(2.0f * x) - 1.0f; }

#define GSTRIDE ((size_t)(H_ / 16) * (H_ / 32) * 512)  // packed elems per gate
#define KTILES  (H_ / 32)

// Accumulators: 2 M-tiles x 4 gates.
struct Acc8 { v8f a00, a01, a02, a03, a10, a11, a12, a13; };

// Dual-M-tile, 4-gate contraction. All 6 loads of an iteration are issued
// up-front as named registers so the compiler can clause them and retire the
// waits incrementally across the 8 WMMAs (no explicit double buffering ->
// no spills).
__device__ __forceinline__ void contract4g(const __bf16* __restrict__ A,
                                           const __bf16* __restrict__ W,
                                           int mb0, int ntile, int lane,
                                           Acc8& acc) {
  size_t tbase = (size_t)ntile * KTILES * 512;
#pragma unroll 2
  for (int kt = 0; kt < KTILES; ++kt) {
    int k0 = kt * 32;
    size_t toff = tbase + (size_t)kt * 512;
    v16bf a0 = load_a(A, H_, mb0, k0, lane);
    v16bf a1 = load_a(A, H_, mb0 + 16, k0, lane);
    v16bf b0 = load_b(W + 0 * GSTRIDE, toff, lane);
    v16bf b1 = load_b(W + 1 * GSTRIDE, toff, lane);
    v16bf b2 = load_b(W + 2 * GSTRIDE, toff, lane);
    v16bf b3 = load_b(W + 3 * GSTRIDE, toff, lane);
    acc.a00 = wmma_bf16(a0, b0, acc.a00);
    acc.a10 = wmma_bf16(a1, b0, acc.a10);
    acc.a01 = wmma_bf16(a0, b1, acc.a01);
    acc.a11 = wmma_bf16(a1, b1, acc.a11);
    acc.a02 = wmma_bf16(a0, b2, acc.a02);
    acc.a12 = wmma_bf16(a1, b2, acc.a12);
    acc.a03 = wmma_bf16(a0, b3, acc.a03);
    acc.a13 = wmma_bf16(a1, b3, acc.a13);
  }
}

// ---------------- utility kernels ----------------

__global__ void zero_kernel(uint32_t* p, size_t n) {
  for (size_t i = (size_t)blockIdx.x * blockDim.x + threadIdx.x; i < n;
       i += (size_t)gridDim.x * blockDim.x)
    p[i] = 0u;
}

// Pack fp32 row-major W[K x Nsrc] into bf16 WMMA-B tiles.
__global__ void pack_w_kernel(const float* __restrict__ src,
                              unsigned short* __restrict__ dst,
                              int K, int Nsrc, int n_tiles) {
  const int ktiles = K / 32;
  size_t total = (size_t)n_tiles * ktiles * 512;
  for (size_t id = (size_t)blockIdx.x * blockDim.x + threadIdx.x; id < total;
       id += (size_t)gridDim.x * blockDim.x) {
    int within = (int)(id & 511);
    size_t tile = id >> 9;
    int lane = within >> 4, i = within & 15;
    int kt = (int)(tile % ktiles);
    int nt = (int)(tile / ktiles);
    int k = kt * 32 + ((i >> 3) << 4) + ((lane >> 4) << 3) + (i & 7);
    int n = nt * 16 + (lane & 15);
    float v = (k < K && n < Nsrc) ? src[(size_t)k * Nsrc + n] : 0.0f;
    dst[id] = f2bf(v);
  }
}

// ---------------- LSTM step (one layer, one timestep) ----------------
// Each wave: 2 M-tiles x 16 hidden cols x 4 gates -> fused in-wave epilogue.
__global__ void __launch_bounds__(128, 1) lstm_step_kernel(
    const __bf16* __restrict__ A1, const __bf16* __restrict__ W1,
    const __bf16* __restrict__ A2, const __bf16* __restrict__ W2,
    const float* __restrict__ xsrc,  // null for layer 2
    const float* __restrict__ wxg, const float* __restrict__ wxi,
    const float* __restrict__ wxf, const float* __restrict__ wxo,
    const float* __restrict__ bg, const float* __restrict__ bi,
    const float* __restrict__ bf, const float* __restrict__ bo,
    float* __restrict__ c,                  // in/out cell state (fp32)
    unsigned short* __restrict__ hout,      // bf16 bits, row-major [B x H]
    int t) {
  const int lane = threadIdx.x & 31;
  const int wave = threadIdx.x >> 5;
  const int ntile = blockIdx.x * 4 + wave;
  const int mb0 = blockIdx.y * 32;          // two 16-row tiles: mb0, mb0+16

  Acc8 acc = {};
  contract4g(A1, W1, mb0, ntile, lane, acc);
  if (A2) contract4g(A2, W2, mb0, ntile, lane, acc);

  const int n = ntile * 16 + (lane & 15);
  float vbg = bg[n], vbi = bi[n], vbf = bf[n], vbo = bo[n];
  float vwg = 0.f, vwi = 0.f, vwf = 0.f, vwo = 0.f;
  if (xsrc) { vwg = wxg[n]; vwi = wxi[n]; vwf = wxf[n]; vwo = wxo[n]; }

#pragma unroll
  for (int mi = 0; mi < 2; ++mi) {
    v8f pg8 = mi ? acc.a10 : acc.a00;
    v8f pi8 = mi ? acc.a11 : acc.a01;
    v8f pf8 = mi ? acc.a12 : acc.a02;
    v8f po8 = mi ? acc.a13 : acc.a03;
    int mb = mb0 + mi * 16;
#pragma unroll
    for (int r = 0; r < 8; ++r) {
      int m = mb + r + ((lane >> 4) << 3);
      float xv = xsrc ? xsrc[(size_t)m * S_ + t] : 0.0f;
      float pg = pg8[r] + vbg + xv * vwg;
      float pi = pi8[r] + vbi + xv * vwi;
      float pf = pf8[r] + vbf + xv * vwf;
      float po = po8[r] + vbo + xv * vwo;
      size_t idx = (size_t)m * H_ + n;
      float cn = tanh_f(pg) * sigm(pi) + c[idx] * sigm(pf);
      c[idx] = cn;
      float hn = tanh_f(cn) * sigm(po);
      hout[idx] = f2bf(hn);
    }
  }
}

// ---------------- final projection: out = h2 @ Wph + bp ----------------
__global__ void __launch_bounds__(256, 1) proj_kernel(
    const __bf16* __restrict__ A,
    const __bf16* __restrict__ Wp,
    const float* __restrict__ bias,
    float* __restrict__ out) {
  const int lane = threadIdx.x & 31;
  const int wave = threadIdx.x >> 5;
  const int ntile = blockIdx.x * 8 + wave;
  if (ntile >= NT_V) return;                 // uniform per wave
  const int mbase = blockIdx.y * 16;

  v8f acc = {};
  size_t tbase = (size_t)ntile * KTILES * 512;
#pragma unroll 2
  for (int kt = 0; kt < KTILES; ++kt) {
    v16bf a = load_a(A, H_, mbase, kt * 32, lane);
    v16bf b = load_b(Wp, tbase + (size_t)kt * 512, lane);
    acc = wmma_bf16(a, b, acc);
  }
  const int n = ntile * 16 + (lane & 15);
  if (n < V_) {
    float bv = bias[n];
#pragma unroll
    for (int r = 0; r < 8; ++r) {
      int m = mbase + r + ((lane >> 4) << 3);
      out[(size_t)m * V_ + n] = acc[r] + bv;
    }
  }
}

// ---------------- host launcher ----------------

extern "C" void kernel_launch(void* const* d_in, const int* in_sizes, int n_in,
                              void* d_out, int out_size, void* d_ws, size_t ws_size,
                              hipStream_t stream) {
  (void)in_sizes; (void)n_in; (void)out_size; (void)ws_size;

  // d_in order: x, then for g in (g,i,f,o): Wgx1, Wgh1, bg1, Wgx2, Wgh2, bg2,
  // then Wph, bp.
  const float* x = (const float*)d_in[0];
  const float *Wx1[4], *Wh1[4], *b1v[4], *Wx2[4], *Wh2[4], *b2v[4];
  for (int g = 0; g < 4; ++g) {
    int base = 1 + 6 * g;
    Wx1[g] = (const float*)d_in[base + 0];
    Wh1[g] = (const float*)d_in[base + 1];
    b1v[g] = (const float*)d_in[base + 2];
    Wx2[g] = (const float*)d_in[base + 3];
    Wh2[g] = (const float*)d_in[base + 4];
    b2v[g] = (const float*)d_in[base + 5];
  }
  const float* Wph = (const float*)d_in[25];
  const float* bp  = (const float*)d_in[26];
  float* out = (float*)d_out;

  // workspace carve-up
  char* ws = (char*)d_ws;
  const size_t hbytes = (size_t)B_ * H_ * 2;   // 512 KB (bf16 h state)
  const size_t cbytes = (size_t)B_ * H_ * 4;   // 1 MB (fp32 c state)
  __bf16* h1[2] = { (__bf16*)(ws + 0 * hbytes), (__bf16*)(ws + 1 * hbytes) };
  __bf16* h2[2] = { (__bf16*)(ws + 2 * hbytes), (__bf16*)(ws + 3 * hbytes) };
  float* c1 = (float*)(ws + 4 * hbytes);
  float* c2 = (float*)(ws + 4 * hbytes + cbytes);
  char* pk = ws + 4 * hbytes + 2 * cbytes;
  unsigned short* pWh1 = (unsigned short*)pk; pk += GSTRIDE * 4 * 2;  // 8 MB
  unsigned short* pWx2 = (unsigned short*)pk; pk += GSTRIDE * 4 * 2;  // 8 MB
  unsigned short* pWh2 = (unsigned short*)pk; pk += GSTRIDE * 4 * 2;  // 8 MB
  unsigned short* pWph = (unsigned short*)pk;                         // ~103 MB

  // zero h/c state (harness poisons ws)
  size_t state_words = (4 * hbytes + 2 * cbytes) / 4;
  zero_kernel<<<1024, 256, 0, stream>>>((uint32_t*)ws, state_words);

  // pack weights fp32 -> bf16 WMMA-B tile layout (L2-resident afterwards)
  for (int g = 0; g < 4; ++g) {
    pack_w_kernel<<<512, 256, 0, stream>>>(Wh1[g], pWh1 + g * GSTRIDE, H_, H_, H_ / 16);
    pack_w_kernel<<<512, 256, 0, stream>>>(Wx2[g], pWx2 + g * GSTRIDE, H_, H_, H_ / 16);
    pack_w_kernel<<<512, 256, 0, stream>>>(Wh2[g], pWh2 + g * GSTRIDE, H_, H_, H_ / 16);
  }
  pack_w_kernel<<<4096, 256, 0, stream>>>(Wph, pWph, H_, V_, NT_V);

  // sequential scan: 2 kernels per timestep, ping-pong h buffers
  dim3 sgrid(H_ / 16 / 4, B_ / 32);  // (16,8), 128 threads = 4 waves
  for (int t = 0; t < S_; ++t) {
    const __bf16* h1r = h1[t & 1];  __bf16* h1w = h1[(t + 1) & 1];
    const __bf16* h2r = h2[t & 1];  __bf16* h2w = h2[(t + 1) & 1];
    // layer 1: recurrent GEMM + rank-1 x-term in epilogue
    lstm_step_kernel<<<sgrid, 128, 0, stream>>>(
        h1r, (const __bf16*)pWh1, nullptr, nullptr, x,
        Wx1[0], Wx1[1], Wx1[2], Wx1[3],
        b1v[0], b1v[1], b1v[2], b1v[3],
        c1, (unsigned short*)h1w, t);
    // layer 2: dual contraction (h1_new @ Wx2 + h2 @ Wh2)
    lstm_step_kernel<<<sgrid, 128, 0, stream>>>(
        h1w, (const __bf16*)pWx2, h2r, (const __bf16*)pWh2, nullptr,
        nullptr, nullptr, nullptr, nullptr,
        b2v[0], b2v[1], b2v[2], b2v[3],
        c2, (unsigned short*)h2w, t);
  }

  // final projection uses h2 written at t = S_-1 -> h2[(S_) & 1] == h2[0]
  dim3 pgrid((NT_V + 7) / 8, B_ / 16);
  proj_kernel<<<pgrid, 256, 0, stream>>>(h2[0], (const __bf16*)pWph, bp, out);
}
